// BiLSTM_CRF_51221779972802
// MI455X (gfx1250) — matrix operand order repaired and checked by reading, not compiled, not run on hardware
//
#include <hip/hip_runtime.h>
#include <hip/hip_bf16.h>

// ---------------- problem constants ----------------
#define VQ 50000
#define EQ 300
#define EP 320      // E padded to multiple of 32 (WMMA K)
#define HQ 256
#define G4H 1024    // 4*H
#define HD 512      // 2*H
#define TQ 9
#define TP 16       // T padded to one WMMA N-tile
#define BQ 64
#define SQ 512
#define SB (SQ*BQ)  // 32768

typedef __attribute__((ext_vector_type(16))) __bf16 bf16x16;
typedef __attribute__((ext_vector_type(8)))  __bf16 bf16x8;
typedef __attribute__((ext_vector_type(8)))  float  f32x8;

union AB16 { bf16x16 v; bf16x8 h[2]; };

__device__ __forceinline__ f32x8 wmma_bf16(bf16x16 a, bf16x16 b, f32x8 c) {
  // D = A(16x32 bf16) * B(32x16 bf16) + C(16x16 f32)
  return __builtin_amdgcn_wmma_f32_16x16x32_bf16(
      /*neg_a=*/false, a, /*neg_b=*/false, b,
      /*c_mod=*/(short)0, c, /*reuse_a=*/false, /*reuse_b=*/false);
}

__device__ __forceinline__ float sigf(float x) { return 1.0f / (1.0f + expf(-x)); }

// ---------------- kernel 1: embedding gather -> bf16 X[SB][EP] ----------------
// x[s][b] = embed[sentence[b][s]]; row = s*B + b
__global__ void k_gather(const int* __restrict__ sentence,
                         const float* __restrict__ embed,
                         __bf16* __restrict__ X) {
  int row = blockIdx.x;            // [0, SB)
  int s = row >> 6;
  int b = row & 63;
  int idx = sentence[b * SQ + s];
  int e = threadIdx.x;             // 320 threads
  float v = (e < EQ) ? embed[(size_t)idx * EQ + e] : 0.0f;
  X[(size_t)row * EP + e] = (__bf16)v;
}

// ---------------- kernel 2a: f32 -> bf16 weight convert (K/cols padding) ------
__global__ void k_cvt(const float* __restrict__ src, __bf16* __restrict__ dst,
                      int cols, int colsPad, int total) {
  int i = blockIdx.x * blockDim.x + threadIdx.x;
  if (i >= total) return;
  int r = i / colsPad, c = i - r * colsPad;
  dst[i] = (c < cols) ? (__bf16)src[r * cols + c] : (__bf16)0.0f;
}

// ---------------- kernel 2b: f32 -> bf16 with row padding (for w_out) ---------
__global__ void k_cvt_rows(const float* __restrict__ src, __bf16* __restrict__ dst,
                           int rows, int cols, int rowsPad) {
  int i = blockIdx.x * blockDim.x + threadIdx.x;
  if (i >= rowsPad * cols) return;
  int r = i / cols;
  dst[i] = (r < rows) ? (__bf16)src[i] : (__bf16)0.0f;
}

// ---------------- kernel 3: input projection GEMM (bf16 WMMA) -----------------
// G[s][n][m] = sum_k X[s*B+m][k] * W[n][k] + bih[n] + bhh[n]   (bf16, transposed
// inner layout so the recurrence reads each gate as one contiguous bf16x8).
// Block = 256 thr (8 waves), block owns M-tile (16 rows), wave owns 8 N-tiles.
__global__ void k_igemm(const __bf16* __restrict__ X,
                        const __bf16* __restrict__ W,
                        const float* __restrict__ bih,
                        const float* __restrict__ bhh,
                        __bf16* __restrict__ Gout) {
  int mTile = blockIdx.x;               // [0, 2048)
  int wave = threadIdx.x >> 5;
  int lane = threadIdx.x & 31;
  int col  = lane & 15;
  int hi   = lane >> 4;

  f32x8 acc[8];
  #pragma unroll
  for (int j = 0; j < 8; ++j) acc[j] = (f32x8){0,0,0,0,0,0,0,0};

  int rowA = mTile * 16 + col;
  const __bf16* xrow = X + (size_t)rowA * EP;

  for (int kt = 0; kt < EP / 32; ++kt) {
    AB16 a;
    a.h[0] = *(const bf16x8*)(xrow + kt * 32 + hi * 16);
    a.h[1] = *(const bf16x8*)(xrow + kt * 32 + hi * 16 + 8);
    #pragma unroll
    for (int j = 0; j < 8; ++j) {
      int n = (wave * 8 + j) * 16 + col;            // B column = W row (contiguous)
      bf16x16 bv = *(const bf16x16*)(W + (size_t)n * EP + kt * 32 + hi * 16);
      acc[j] = wmma_bf16(a.v, bv, acc[j]);
    }
  }

  // store transposed: all 8 accumulator rows of one lane are consecutive batch
  // indices of the same time step -> one contiguous bf16x8 store per tile.
  int sIdx = mTile >> 2;                 // 4 M-tiles per time step (B=64)
  int mlo  = (mTile & 3) * 16 + hi * 8;  // batch offset of acc row r=0
  #pragma unroll
  for (int j = 0; j < 8; ++j) {
    int n = (wave * 8 + j) * 16 + col;
    float bias = bih[n] + bhh[n];
    bf16x8 out;
    #pragma unroll
    for (int r = 0; r < 8; ++r) out[r] = (__bf16)(acc[j][r] + bias);
    *(bf16x8*)(Gout + ((size_t)sIdx * G4H + n) * BQ + mlo) = out;
  }
}

// ---------------- kernel 4: LSTM recurrence (bf16 WMMA, fused gates) ----------
// 8 blocks: dir = blockIdx.x>>2, batch M-tile = blockIdx.x&3. 512 thr = 16 waves.
// Wave w owns gate tiles {i,f,g,o} for hidden block j = w*16..w*16+15 -> the 4
// accumulators align lane/register-wise with cell state c (8 f32/lane).
__global__ void __launch_bounds__(512)
k_rec(const __bf16* __restrict__ Gf, const __bf16* __restrict__ Gb,
      const __bf16* __restrict__ WhhF, const __bf16* __restrict__ WhhB,
      __bf16* __restrict__ hcatbf) {
  int dir = blockIdx.x >> 2;
  int mt  = blockIdx.x & 3;
  const __bf16* G = dir ? Gb : Gf;
  const __bf16* W = dir ? WhhB : WhhF;

  __shared__ __align__(32) __bf16 hT[16][HQ];   // 8 KB h tile (bf16)

  int tid  = threadIdx.x;
  int wave = tid >> 5;
  int lane = tid & 31;
  int col  = lane & 15;
  int hi   = lane >> 4;

  for (int i = tid; i < 16 * HQ; i += 512) ((__bf16*)hT)[i] = (__bf16)0.0f;
  __syncthreads();

  float c[8];
  #pragma unroll
  for (int r = 0; r < 8; ++r) c[r] = 0.0f;

  int mbase = mt * 16;
  int j = wave * 16 + col;                       // hidden index this lane owns
  int moff = mbase + hi * 8;                     // batch offset of acc row r=0

  for (int t = 0; t < SQ; ++t) {
    int s = dir ? (SQ - 1 - t) : t;

    // prefetch this step's gate vectors so they overlap the WMMA chain
    const __bf16* gp = G + ((size_t)s * G4H + j) * BQ + moff;
    __builtin_prefetch(gp, 0, 3);
    __builtin_prefetch(gp + (size_t)1 * HQ * BQ, 0, 3);
    __builtin_prefetch(gp + (size_t)2 * HQ * BQ, 0, 3);
    __builtin_prefetch(gp + (size_t)3 * HQ * BQ, 0, 3);

    f32x8 acc0 = (f32x8){0,0,0,0,0,0,0,0};
    f32x8 acc1 = acc0, acc2 = acc0, acc3 = acc0;

    #pragma unroll
    for (int kt = 0; kt < HQ / 32; ++kt) {
      AB16 a;
      a.h[0] = *(const bf16x8*)&hT[col][kt * 32 + hi * 16];
      a.h[1] = *(const bf16x8*)&hT[col][kt * 32 + hi * 16 + 8];
      const __bf16* wk = W + (size_t)(kt * 32 + hi * 16);
      acc0 = wmma_bf16(a.v, *(const bf16x16*)(wk + (size_t)(0 * HQ + j) * HQ), acc0);
      acc1 = wmma_bf16(a.v, *(const bf16x16*)(wk + (size_t)(1 * HQ + j) * HQ), acc1);
      acc2 = wmma_bf16(a.v, *(const bf16x16*)(wk + (size_t)(2 * HQ + j) * HQ), acc2);
      acc3 = wmma_bf16(a.v, *(const bf16x16*)(wk + (size_t)(3 * HQ + j) * HQ), acc3);
    }

    // gate pre-activations: one contiguous 16B load per gate (transposed G)
    bf16x8 gv0 = *(const bf16x8*)(gp);
    bf16x8 gv1 = *(const bf16x8*)(gp + (size_t)1 * HQ * BQ);
    bf16x8 gv2 = *(const bf16x8*)(gp + (size_t)2 * HQ * BQ);
    bf16x8 gv3 = *(const bf16x8*)(gp + (size_t)3 * HQ * BQ);

    __syncthreads();   // all LDS h reads for this step complete

    #pragma unroll
    for (int r = 0; r < 8; ++r) {
      int ml = hi * 8 + r;
      int m  = mbase + ml;
      float gi = acc0[r] + (float)gv0[r];
      float gf = acc1[r] + (float)gv1[r];
      float gg = acc2[r] + (float)gv2[r];
      float go = acc3[r] + (float)gv3[r];
      float cn = sigf(gf) * c[r] + sigf(gi) * tanhf(gg);
      c[r] = cn;
      float h = sigf(go) * tanhf(cn);
      hT[ml][j] = (__bf16)h;
      hcatbf[(size_t)(s * BQ + m) * HD + dir * HQ + j] = (__bf16)h;
    }
    __syncthreads();   // h writes visible before next step's reads
  }
}

// ---------------- kernel 5: emission projection (bf16 WMMA, N=16 padded) ------
// emis[row][t] = hcat[row][:] . w_out[t][:] + b_out[t], t < 9.
// Block = 256 thr (8 waves), each wave owns one M-tile, single N-tile, K=512.
__global__ void k_emis(const __bf16* __restrict__ hcatbf,
                       const __bf16* __restrict__ Wout,   // [16][512] bf16, rows 9..15 zero
                       const float* __restrict__ b_out,
                       float* __restrict__ emis) {
  int mTile = blockIdx.x * 8 + (threadIdx.x >> 5);   // [0, 2048)
  int lane = threadIdx.x & 31;
  int col  = lane & 15;
  int hi   = lane >> 4;

  f32x8 acc = (f32x8){0,0,0,0,0,0,0,0};
  const __bf16* arow = hcatbf + (size_t)(mTile * 16 + col) * HD;
  const __bf16* brow = Wout   + (size_t)col * HD;

  #pragma unroll
  for (int kt = 0; kt < HD / 32; ++kt) {
    AB16 a;
    a.h[0] = *(const bf16x8*)(arow + kt * 32 + hi * 16);
    a.h[1] = *(const bf16x8*)(arow + kt * 32 + hi * 16 + 8);
    bf16x16 bv = *(const bf16x16*)(brow + kt * 32 + hi * 16);
    acc = wmma_bf16(a.v, bv, acc);
  }

  if (col < TQ) {
    float bo = b_out[col];
    #pragma unroll
    for (int r = 0; r < 8; ++r) {
      int row = mTile * 16 + hi * 8 + r;
      emis[(size_t)row * TQ + col] = acc[r] + bo;
    }
  }
}

// ---------------- kernel 6: CRF score + forward algorithm + loss --------------
__global__ void k_crf(const float* __restrict__ emis,
                      const int* __restrict__ tags,
                      const int* __restrict__ mask,
                      const float* __restrict__ st,
                      const float* __restrict__ en,
                      const float* __restrict__ tr,
                      float* __restrict__ out) {
  __shared__ float red[BQ];
  int b = threadIdx.x;                 // 64 threads

  int len = 0;
  for (int s = 0; s < SQ; ++s) len += mask[b * SQ + s];

  int tprev = tags[b * SQ + 0];
  float score = st[tprev] + emis[(size_t)(0 * BQ + b) * TQ + tprev];

  float alpha[TQ];
  #pragma unroll
  for (int t = 0; t < TQ; ++t) alpha[t] = st[t] + emis[(size_t)(0 * BQ + b) * TQ + t];

  for (int s = 1; s < SQ; ++s) {
    int tcur = tags[b * SQ + s];
    const float* em = emis + (size_t)(s * BQ + b) * TQ;
    if (mask[b * SQ + s]) {
      score += tr[tprev * TQ + tcur] + em[tcur];
      float nxt[TQ];
      #pragma unroll
      for (int tp = 0; tp < TQ; ++tp) {
        float mx = -1e30f;
        float v[TQ];
        #pragma unroll
        for (int t0 = 0; t0 < TQ; ++t0) {
          v[t0] = alpha[t0] + tr[t0 * TQ + tp] + em[tp];
          mx = fmaxf(mx, v[t0]);
        }
        float sme = 0.0f;
        #pragma unroll
        for (int t0 = 0; t0 < TQ; ++t0) sme += expf(v[t0] - mx);
        nxt[tp] = mx + logf(sme);
      }
      #pragma unroll
      for (int tp = 0; tp < TQ; ++tp) alpha[tp] = nxt[tp];
    }
    tprev = tcur;
  }

  int last_tag = tags[b * SQ + (len - 1)];
  score += en[last_tag];

  float mx = -1e30f;
  #pragma unroll
  for (int t = 0; t < TQ; ++t) mx = fmaxf(mx, alpha[t] + en[t]);
  float sme = 0.0f;
  #pragma unroll
  for (int t = 0; t < TQ; ++t) sme += expf(alpha[t] + en[t] - mx);
  float logZ = mx + logf(sme);

  red[b] = score - logZ;
  __syncthreads();
  if (b == 0) {
    float s = 0.0f;
    for (int i = 0; i < BQ; ++i) s += red[i];
    out[0] = -s / (float)BQ;
  }
}

// ---------------- host orchestration ----------------
extern "C" void kernel_launch(void* const* d_in, const int* in_sizes, int n_in,
                              void* d_out, int out_size, void* d_ws, size_t ws_size,
                              hipStream_t stream) {
  const int*   sentence = (const int*)  d_in[0];
  const int*   tags     = (const int*)  d_in[1];
  const float* embed    = (const float*)d_in[2];
  const float* wih_f    = (const float*)d_in[3];
  const float* whh_f    = (const float*)d_in[4];
  const float* bih_f    = (const float*)d_in[5];
  const float* bhh_f    = (const float*)d_in[6];
  const float* wih_b    = (const float*)d_in[7];
  const float* whh_b    = (const float*)d_in[8];
  const float* bih_b    = (const float*)d_in[9];
  const float* bhh_b    = (const float*)d_in[10];
  const float* w_out    = (const float*)d_in[11];
  const float* b_out    = (const float*)d_in[12];
  const float* st       = (const float*)d_in[13];
  const float* en       = (const float*)d_in[14];
  const float* tr       = (const float*)d_in[15];
  const int*   mask     = (const int*)  d_in[16];

  // workspace carve-out (256B aligned slabs)
  char* p = (char*)d_ws;
  auto carve = [&](size_t bytes) -> void* {
    void* r = (void*)p;
    p += (bytes + 255) & ~(size_t)255;
    return r;
  };
  __bf16* Xbf    = (__bf16*)carve((size_t)SB * EP * 2);       // 20 MB
  __bf16* WihFbf = (__bf16*)carve((size_t)G4H * EP * 2);
  __bf16* WihBbf = (__bf16*)carve((size_t)G4H * EP * 2);
  __bf16* WhhFbf = (__bf16*)carve((size_t)G4H * HQ * 2);
  __bf16* WhhBbf = (__bf16*)carve((size_t)G4H * HQ * 2);
  __bf16* WoutBf = (__bf16*)carve((size_t)TP * HD * 2);
  __bf16* Gf     = (__bf16*)carve((size_t)SB * G4H * 2);      // 64 MB, [s][n][m]
  __bf16* Gb     = (__bf16*)carve((size_t)SB * G4H * 2);      // 64 MB, [s][n][m]
  __bf16* hcatbf = (__bf16*)carve((size_t)SB * HD * 2);       // 32 MB
  float*  emis   = (float*) carve((size_t)SB * TQ * 4);
  (void)ws_size; (void)in_sizes; (void)n_in; (void)out_size;

  // 1. gather + bf16 convert of embeddings
  k_gather<<<SB, EP, 0, stream>>>(sentence, embed, Xbf);

  // 2. weight conversions
  {
    int tot = G4H * EP;
    k_cvt<<<(tot + 255) / 256, 256, 0, stream>>>(wih_f, WihFbf, EQ, EP, tot);
    k_cvt<<<(tot + 255) / 256, 256, 0, stream>>>(wih_b, WihBbf, EQ, EP, tot);
    int tot2 = G4H * HQ;
    k_cvt<<<(tot2 + 255) / 256, 256, 0, stream>>>(whh_f, WhhFbf, HQ, HQ, tot2);
    k_cvt<<<(tot2 + 255) / 256, 256, 0, stream>>>(whh_b, WhhBbf, HQ, HQ, tot2);
    int tot3 = TP * HD;
    k_cvt_rows<<<(tot3 + 255) / 256, 256, 0, stream>>>(w_out, WoutBf, TQ, HD, TP);
  }

  // 3. input projection GEMMs (bias folded in, transposed store)
  k_igemm<<<SB / 16, 256, 0, stream>>>(Xbf, WihFbf, bih_f, bhh_f, Gf);
  k_igemm<<<SB / 16, 256, 0, stream>>>(Xbf, WihBbf, bih_b, bhh_b, Gb);

  // 4. bidirectional LSTM recurrence (2 dirs x 4 batch tiles)
  k_rec<<<8, 512, 0, stream>>>(Gf, Gb, WhhFbf, WhhBbf, hcatbf);

  // 5. emissions via WMMA (N padded to 16)
  k_emis<<<SB / 16 / 8, 256, 0, stream>>>(hcatbf, WoutBf, b_out, emis);

  // 6. CRF loss
  k_crf<<<1, BQ, 0, stream>>>(emis, tags, mask, st, en, tr, (float*)d_out);
}
